// PackedW2A16Linear_62723702391220
// MI455X (gfx1250) — matrix operand order repaired and checked by reading, not compiled, not run on hardware
//
#include <hip/hip_runtime.h>
#include <hip/hip_bf16.h>

// ---- problem dims (from reference setup_inputs) ----
#define DIM_M 8192   // B*S = 4*2048
#define DIM_K 4096
#define DIM_N 4096
#define GRP   64     // quant group size
#define NGRP  (DIM_K / GRP)   // 64

// ---- tiling ----
#define BLOCK_M 256
#define BLOCK_N 128
#define BLOCK_K 64          // one quant group per stage
#define WAVE_M  64
#define WAVE_N  64
#define LDA_PAD 72          // halves per LDS row (64 + 8 pad)

typedef __attribute__((ext_vector_type(16))) _Float16 v16h;
typedef __attribute__((ext_vector_type(8)))  _Float16 half8;
typedef __attribute__((ext_vector_type(4)))  _Float16 half4;
typedef __attribute__((ext_vector_type(8)))  float    v8f;

static __device__ __forceinline__ v16h cat16(half8 lo, half8 hi) {
    v16h r;
#pragma unroll
    for (int i = 0; i < 8; ++i) { r[i] = lo[i]; r[i + 8] = hi[i]; }
    return r;
}

__global__ __launch_bounds__(256)
void PackedW2A16Linear_62723702391220_kernel(const float* __restrict__ x,
                                             const int*   __restrict__ wq,
                                             const float* __restrict__ scale,
                                             const float* __restrict__ zp,
                                             const float* __restrict__ bias,
                                             float*       __restrict__ out) {
    __shared__ _Float16 As[BLOCK_M * LDA_PAD];   // 36,864 B
    __shared__ _Float16 Bs[BLOCK_N * LDA_PAD];   // 18,432 B

    const int tid  = threadIdx.x;
    const int m0   = blockIdx.y * BLOCK_M;
    const int n0   = blockIdx.x * BLOCK_N;

    const int tcol = tid & 15;   // float4/int4 column within 64-wide K chunk
    const int trow = tid >> 4;   // 0..15

    const int lane = tid & 31;
    const int lm   = lane & 15;
    const int l16  = lane >> 4;          // 0 or 1 (half-wave)
    const int wave = tid >> 5;           // 0..7
    const int wm   = wave >> 1;          // 0..3 -> M offset wm*64
    const int wn   = wave & 1;           // 0..1 -> N offset wn*64

    v8f acc[4][4];
#pragma unroll
    for (int mi = 0; mi < 4; ++mi)
#pragma unroll
        for (int ni = 0; ni < 4; ++ni)
            acc[mi][ni] = (v8f){0.f, 0.f, 0.f, 0.f, 0.f, 0.f, 0.f, 0.f};

    for (int kt = 0; kt < DIM_K / BLOCK_K; ++kt) {
        const int k0 = kt * BLOCK_K;
        const int g  = kt;               // group index == stage index

        // hint next stage into cache (global_prefetch_b8)
        if (kt + 1 < DIM_K / BLOCK_K) {
            __builtin_prefetch(x  + (size_t)(m0 + trow) * DIM_K + k0 + BLOCK_K + tcol * 4, 0, 1);
            __builtin_prefetch(wq + (size_t)(n0 + trow) * DIM_K + k0 + BLOCK_K + tcol * 4, 0, 1);
        }

        // ---- stage A: x[256 x 64] fp32 -> f16 in LDS ----
#pragma unroll
        for (int p = 0; p < 16; ++p) {
            const int r = trow + p * 16;                       // 0..255
            const float4 v = *(const float4*)(x + (size_t)(m0 + r) * DIM_K + k0 + tcol * 4);
            half4 h = {(_Float16)v.x, (_Float16)v.y, (_Float16)v.z, (_Float16)v.w};
            *(half4*)(&As[r * LDA_PAD + tcol * 4]) = h;
        }

        // ---- stage B: dequant W[128 x 64] -> f16 in LDS (row-major [n][k]) ----
#pragma unroll
        for (int p = 0; p < 8; ++p) {
            const int nr = trow + p * 16;                      // 0..127
            const int n  = n0 + nr;
            const float sc = scale[(size_t)n * NGRP + g];
            const float z  = zp[(size_t)n * NGRP + g];
            const int4 q = *(const int4*)(wq + (size_t)n * DIM_K + k0 + tcol * 4);
            half4 h = {(_Float16)(((float)q.x - z) * sc),
                       (_Float16)(((float)q.y - z) * sc),
                       (_Float16)(((float)q.z - z) * sc),
                       (_Float16)(((float)q.w - z) * sc)};
            *(half4*)(&Bs[nr * LDA_PAD + tcol * 4]) = h;
        }

        __syncthreads();

        // ---- compute: 2 k-steps of 32, 4x4 WMMA tiles per wave ----
#pragma unroll
        for (int kk = 0; kk < BLOCK_K; kk += 32) {
            v16h a[4], b[4];
#pragma unroll
            for (int mi = 0; mi < 4; ++mi) {
                // A 16x32 f16 layout: lane half (l16) selects K+8 shift;
                // halves 0-7 = K[c0..c0+7], halves 8-15 = K[c0+16..c0+23]
                const int row = wm * WAVE_M + mi * 16 + lm;
                const int c0  = kk + l16 * 8;
                half8 lo = *(const half8*)(&As[row * LDA_PAD + c0]);
                half8 hi = *(const half8*)(&As[row * LDA_PAD + c0 + 16]);
                a[mi] = cat16(lo, hi);
            }
#pragma unroll
            for (int ni = 0; ni < 4; ++ni) {
                // B 32x16 f16 layout: lane = column n, lane half (l16) selects K+8 shift;
                // halves 0-7 = K[c0..c0+7], halves 8-15 = K[c0+16..c0+23]
                const int nrow = wn * WAVE_N + ni * 16 + lm;
                const int c0   = kk + l16 * 8;
                half8 lo = *(const half8*)(&Bs[nrow * LDA_PAD + c0]);
                half8 hi = *(const half8*)(&Bs[nrow * LDA_PAD + c0 + 16]);
                b[ni] = cat16(lo, hi);
            }
#pragma unroll
            for (int mi = 0; mi < 4; ++mi)
#pragma unroll
                for (int ni = 0; ni < 4; ++ni)
                    acc[mi][ni] = __builtin_amdgcn_wmma_f32_16x16x32_f16(
                        /*neg_a=*/false, a[mi], /*neg_b=*/false, b[ni],
                        /*c_mod=*/(short)0, acc[mi][ni],
                        /*reuse_a=*/false, /*reuse_b=*/false);
        }

        __syncthreads();
    }

    // ---- epilogue: C/D layout -> out[m][n] + bias[n] ----
#pragma unroll
    for (int ni = 0; ni < 4; ++ni) {
        const int n  = n0 + wn * WAVE_N + ni * 16 + lm;
        const float bv = bias[n];
#pragma unroll
        for (int mi = 0; mi < 4; ++mi) {
            const int mbase = m0 + wm * WAVE_M + mi * 16 + l16 * 8;
#pragma unroll
            for (int i = 0; i < 8; ++i)
                out[(size_t)(mbase + i) * DIM_N + n] = acc[mi][ni][i] + bv;
        }
    }
}

extern "C" void kernel_launch(void* const* d_in, const int* in_sizes, int n_in,
                              void* d_out, int out_size, void* d_ws, size_t ws_size,
                              hipStream_t stream) {
    const float* x     = (const float*)d_in[0];
    const int*   wq    = (const int*)d_in[1];
    const float* scale = (const float*)d_in[2];
    const float* zp    = (const float*)d_in[3];
    const float* bias  = (const float*)d_in[4];
    float*       out   = (float*)d_out;

    dim3 grid(DIM_N / BLOCK_N, DIM_M / BLOCK_M);   // 32 x 32
    dim3 block(256);                                // 8 wave32s
    PackedW2A16Linear_62723702391220_kernel<<<grid, block, 0, stream>>>(
        x, wq, scale, zp, bias, out);
}
